// SAGE_39565238731349
// MI455X (gfx1250) — matrix operand order repaired and checked by reading, not compiled, not run on hardware
//
#include <hip/hip_runtime.h>
#include <hip/hip_bf16.h>

// ---------------------------------------------------------------------------
// GraphSAGE (3x SAGEConv + BN + LeakyReLU) -> add-pool -> BN -> FC
// N=100000 nodes x 128 feats, E=1600000 edges, G=1024 graphs, out 1024x64.
// ---------------------------------------------------------------------------

#define NNODES 100000
#define NEDGES 1600000
#define NGRAPH 1024
#define HDIM   128
#define BN_EPS 1e-5f
#define SLOPE  0.2f

typedef __attribute__((ext_vector_type(16))) __bf16 v16bf;
typedef __attribute__((ext_vector_type(8)))  __bf16 v8bf;
typedef __attribute__((ext_vector_type(8)))  float  v8f;

__device__ __forceinline__ __bf16 f2bf(float f) {
    unsigned u = __builtin_bit_cast(unsigned, f);
    unsigned r = u + 0x7FFFu + ((u >> 16) & 1u);   // round-to-nearest-even
    unsigned short h = (unsigned short)(r >> 16);
    return __builtin_bit_cast(__bf16, h);
}

// --------------------------- degree / inv-degree ---------------------------
__global__ void deg_kernel(const int* __restrict__ ei, float* __restrict__ deg) {
    int e = blockIdx.x * 256 + threadIdx.x;
    if (e < NEDGES) atomicAdd(&deg[ei[NEDGES + e]], 1.0f);
}

__global__ void invdeg_kernel(float* __restrict__ deg) {
    int i = blockIdx.x * 256 + threadIdx.x;
    if (i < NNODES) {
        float d = deg[i];
        deg[i] = d > 0.0f ? 1.0f / d : 0.0f;
    }
}

// ------------------------- edge scatter aggregation ------------------------
// one wave per edge: float4 gather of h[src], 4 atomic f32 adds per lane
__global__ void agg_kernel(const float* __restrict__ h, const int* __restrict__ ei,
                           float* __restrict__ agg) {
    int gid  = blockIdx.x * 256 + threadIdx.x;
    int e    = gid >> 5;
    int lane = gid & 31;
    if (e >= NEDGES) return;
    int src = ei[e];
    int dst = ei[NEDGES + e];
    const float4* hs = (const float4*)(h + (size_t)src * HDIM);
    float*        ad = agg + (size_t)dst * HDIM;
    float4 v = hs[lane];
    int c = lane * 4;
    atomicAdd(ad + c + 0, v.x);
    atomicAdd(ad + c + 1, v.y);
    atomicAdd(ad + c + 2, v.z);
    atomicAdd(ad + c + 3, v.w);
}

// ------------------------------ WMMA GEMM ----------------------------------
// out(in-place on aggio) = (aggio * inv_deg) @ Wl + h @ Wr + bl
// block: 256 threads = 8 waves, each wave does a 16-row slab x 128 cols.
// Weights staged in LDS as bf16, padded row stride (16B aligned, fewer bank
// conflicts). bf16 WMMA 16x16x32, f32 accumulate.
#define LDSSTR 136                       // halves per row (272B, 16B aligned)
#define NSLAB  (NNODES / 16)             // 6250

__device__ __forceinline__ v16bf load_a_frag(const float* __restrict__ p, float s) {
    // p points at rowBase + 32*kstep + (lane>=16 ? 8 : 0)
    float4 a0 = ((const float4*)p)[0];   // K +0..3
    float4 a1 = ((const float4*)p)[1];   // K +4..7
    float4 a2 = ((const float4*)p)[4];   // K +16..19
    float4 a3 = ((const float4*)p)[5];   // K +20..23
    v16bf f;
    f[0]  = f2bf(a0.x * s); f[1]  = f2bf(a0.y * s);
    f[2]  = f2bf(a0.z * s); f[3]  = f2bf(a0.w * s);
    f[4]  = f2bf(a1.x * s); f[5]  = f2bf(a1.y * s);
    f[6]  = f2bf(a1.z * s); f[7]  = f2bf(a1.w * s);
    f[8]  = f2bf(a2.x * s); f[9]  = f2bf(a2.y * s);
    f[10] = f2bf(a2.z * s); f[11] = f2bf(a2.w * s);
    f[12] = f2bf(a3.x * s); f[13] = f2bf(a3.y * s);
    f[14] = f2bf(a3.z * s); f[15] = f2bf(a3.w * s);
    return f;
}

__device__ __forceinline__ v16bf load_b_frag(const __bf16* __restrict__ p) {
    v16bf b;
#pragma unroll
    for (int i = 0; i < 8; ++i) { b[i] = p[i * LDSSTR]; b[8 + i] = p[(16 + i) * LDSSTR]; }
    return b;
}

__global__ void __launch_bounds__(256)
gemm_kernel(float* __restrict__ aggio, const float* __restrict__ h,
            const float* __restrict__ invdeg,
            const float* __restrict__ Wl, const float* __restrict__ bl,
            const float* __restrict__ Wr) {
    __shared__ __bf16 ldsW[2][HDIM * LDSSTR];

    int tid = threadIdx.x;
    // stage both 128x128 weight matrices into LDS as bf16 (coalesced)
#pragma unroll 4
    for (int j = 0; j < 64; ++j) {
        int idx = j * 256 + tid;         // 0..16383
        int r = idx >> 7, c = idx & 127;
        ldsW[0][r * LDSSTR + c] = f2bf(Wl[idx]);
        ldsW[1][r * LDSSTR + c] = f2bf(Wr[idx]);
    }
    __syncthreads();

    int wave = tid >> 5;
    int lane = tid & 31;
    int slab = blockIdx.x * 8 + wave;    // wave-uniform guard (EXEC all-1 for WMMA)
    if (slab >= NSLAB) return;

    int m15    = lane & 15;
    int hiHalf = (lane & 16) ? 8 : 0;
    size_t row = (size_t)slab * 16 + m15;
    const float* aRow = aggio + row * HDIM;
    const float* hRow = h     + row * HDIM;
    float scale = invdeg[row];

    v8f acc[8];
#pragma unroll
    for (int ct = 0; ct < 8; ++ct) {
        float b = bl[ct * 16 + m15];     // per-column bias, broadcast across M
#pragma unroll
        for (int r = 0; r < 8; ++r) acc[ct][r] = b;
    }

#pragma unroll
    for (int s = 0; s < 4; ++s) {        // K steps of 32
        int koff = s * 32 + hiHalf;
        v16bf aA = load_a_frag(aRow + koff, scale);
        v16bf aH = load_a_frag(hRow + koff, 1.0f);
        const __bf16* wl = &ldsW[0][(s * 32 + hiHalf) * LDSSTR + m15];
        const __bf16* wr = &ldsW[1][(s * 32 + hiHalf) * LDSSTR + m15];
#pragma unroll
        for (int ct = 0; ct < 8; ++ct) {
            v16bf bA = load_b_frag(wl + ct * 16);
            acc[ct] = __builtin_amdgcn_wmma_f32_16x16x32_bf16(
                false, aA, false, bA, (short)0, acc[ct], false, false);
            v16bf bH = load_b_frag(wr + ct * 16);
            acc[ct] = __builtin_amdgcn_wmma_f32_16x16x32_bf16(
                false, aH, false, bH, (short)0, acc[ct], false, false);
        }
    }

    // store 16x128 slab (in-place on aggio; this wave already consumed its rows)
    float* outBase = aggio + ((size_t)slab * 16 + hiHalf) * HDIM;
#pragma unroll
    for (int ct = 0; ct < 8; ++ct) {
        int col = ct * 16 + m15;
#pragma unroll
        for (int r = 0; r < 8; ++r)
            outBase[(size_t)r * HDIM + col] = acc[ct][r];
    }
}

// --------------------------- BatchNorm stats -------------------------------
__global__ void stats_kernel(const float* __restrict__ h, float* __restrict__ sum,
                             float* __restrict__ sq, int M, int rowsPerBlock) {
    __shared__ float ssum[256], ssq[256];
    int col  = threadIdx.x & 127;
    int half = threadIdx.x >> 7;
    int base = blockIdx.x * rowsPerBlock + half;
    float s = 0.0f, q = 0.0f;
    for (int i = 0; i < rowsPerBlock; i += 2) {
        int r = base + i;
        if (r < M) {
            float v = h[(size_t)r * HDIM + col];
            s += v; q += v * v;
        }
    }
    ssum[threadIdx.x] = s; ssq[threadIdx.x] = q;
    __syncthreads();
    if (half == 0) {
        atomicAdd(&sum[col], ssum[col] + ssum[col + 128]);
        atomicAdd(&sq[col],  ssq[col]  + ssq[col + 128]);
    }
}

__global__ void bnfin_kernel(const float* __restrict__ sum, const float* __restrict__ sq,
                             float* __restrict__ mu, float* __restrict__ rstd, float invM) {
    int c = threadIdx.x;
    float m = sum[c] * invM;
    float v = sq[c] * invM - m * m;
    mu[c]   = m;
    rstd[c] = rsqrtf(v + BN_EPS);
}

__global__ void bnapply_kernel(float* __restrict__ h, const float* __restrict__ mu,
                               const float* __restrict__ rstd,
                               const float* __restrict__ gamma,
                               const float* __restrict__ beta) {
    int i = blockIdx.x * 256 + threadIdx.x;   // total = NNODES*128 = 12.8M
    int c = i & 127;
    float v = h[i];
    v = gamma[c] * (v - mu[c]) * rstd[c] + beta[c];
    h[i] = v > 0.0f ? v : SLOPE * v;          // leaky relu
}

// ----------------------------- global add pool -----------------------------
__global__ void pool_kernel(const float* __restrict__ h, const int* __restrict__ batch,
                            float* __restrict__ pooled) {
    int i = blockIdx.x * 256 + threadIdx.x;
    if (i >= NNODES * HDIM) return;
    int node = i >> 7, c = i & 127;
    atomicAdd(&pooled[batch[node] * HDIM + c], h[i]);
}

// --------------------------- final BN + FC ---------------------------------
__global__ void final_kernel(const float* __restrict__ pooled,
                             const float* __restrict__ mu, const float* __restrict__ rstd,
                             const float* __restrict__ g2, const float* __restrict__ b2,
                             const float* __restrict__ fcW, const float* __restrict__ fcb,
                             float* __restrict__ out) {
    int i = blockIdx.x * 256 + threadIdx.x;   // 1024*64
    if (i >= NGRAPH * 64) return;
    int g = i >> 6, j = i & 63;
    const float* pr = pooled + g * HDIM;
    float acc = fcb[j];
#pragma unroll 4
    for (int k = 0; k < HDIM; ++k) {
        float v = g2[k] * (pr[k] - mu[k]) * rstd[k] + b2[k];
        acc += v * fcW[k * 64 + j];
    }
    out[i] = acc;
}

// ---------------------------------------------------------------------------
extern "C" void kernel_launch(void* const* d_in, const int* in_sizes, int n_in,
                              void* d_out, int out_size, void* d_ws, size_t ws_size,
                              hipStream_t stream) {
    const float* x    = (const float*)d_in[0];
    const int*   ei   = (const int*)d_in[1];
    const int*   bat  = (const int*)d_in[2];
    const float* Wl   = (const float*)d_in[3];
    const float* bl   = (const float*)d_in[4];
    const float* Wr   = (const float*)d_in[5];
    const float* bng  = (const float*)d_in[6];
    const float* bnb  = (const float*)d_in[7];
    const float* bn2g = (const float*)d_in[8];
    const float* bn2b = (const float*)d_in[9];
    const float* fcW  = (const float*)d_in[10];
    const float* fcb  = (const float*)d_in[11];

    const size_t N128 = (size_t)NNODES * HDIM;   // 12.8M floats
    float* bufA   = (float*)d_ws;
    float* bufB   = bufA + N128;
    float* invdeg = bufB + N128;
    float* stats  = invdeg + ((NNODES + 63) / 64) * 64;
    float* sumv   = stats;
    float* sqv    = stats + 128;
    float* muv    = stats + 256;
    float* rstdv  = stats + 384;
    float* pooled = stats + 512;

    // h0 = x
    hipMemcpyAsync(bufA, x, N128 * sizeof(float), hipMemcpyDeviceToDevice, stream);

    // in-degree -> 1/deg (in place)
    hipMemsetAsync(invdeg, 0, NNODES * sizeof(float), stream);
    deg_kernel<<<(NEDGES + 255) / 256, 256, 0, stream>>>(ei, invdeg);
    invdeg_kernel<<<(NNODES + 255) / 256, 256, 0, stream>>>(invdeg);

    for (int layer = 0; layer < 3; ++layer) {
        // agg = segment_sum(h[src], dst)
        hipMemsetAsync(bufB, 0, N128 * sizeof(float), stream);
        agg_kernel<<<(NEDGES * 32) / 256, 256, 0, stream>>>(bufA, ei, bufB);
        // h' = (agg*invdeg) @ Wl + bl + h @ Wr   (in place on bufB, bf16 WMMA)
        gemm_kernel<<<(NSLAB + 7) / 8, 256, 0, stream>>>(
            bufB, bufA, invdeg, Wl + layer * HDIM * HDIM, bl + layer * HDIM,
            Wr + layer * HDIM * HDIM);
        // batch-norm (training stats) + leaky relu, in place
        hipMemsetAsync(sumv, 0, 256 * sizeof(float), stream);
        stats_kernel<<<(NNODES + 511) / 512, 256, 0, stream>>>(bufB, sumv, sqv, NNODES, 512);
        bnfin_kernel<<<1, 128, 0, stream>>>(sumv, sqv, muv, rstdv, 1.0f / NNODES);
        bnapply_kernel<<<(int)(N128 / 256), 256, 0, stream>>>(
            bufB, muv, rstdv, bng + layer * HDIM, bnb + layer * HDIM);
        float* t = bufA; bufA = bufB; bufB = t;
    }

    // global add pool
    hipMemsetAsync(pooled, 0, (size_t)NGRAPH * HDIM * sizeof(float), stream);
    pool_kernel<<<(NNODES * HDIM + 255) / 256, 256, 0, stream>>>(bufA, bat, pooled);

    // BN over pooled, then FC -> out
    hipMemsetAsync(sumv, 0, 256 * sizeof(float), stream);
    stats_kernel<<<NGRAPH / 64, 256, 0, stream>>>(pooled, sumv, sqv, NGRAPH, 64);
    bnfin_kernel<<<1, 128, 0, stream>>>(sumv, sqv, muv, rstdv, 1.0f / NGRAPH);
    final_kernel<<<(NGRAPH * 64 + 255) / 256, 256, 0, stream>>>(
        pooled, muv, rstdv, bn2g, bn2b, fcW, fcb, (float*)d_out);
}
